// SAttendedSimpleNet_4286377362011
// MI455X (gfx1250) — matrix operand order repaired and checked by reading, not compiled, not run on hardware
//
#include <hip/hip_runtime.h>
#include <hip/hip_bf16.h>

typedef __attribute__((ext_vector_type(16))) _Float16 v16h;
typedef __attribute__((ext_vector_type(8)))  float    v8f;

#define T_LEN 256
#define BATCH 128
#define EMB   512
#define HID   512
#define NH    8
#define ASZ   64
#define G4H   2048          // 4*HID
#define TBTOT (T_LEN*BATCH) // 32768

// ---------------------------------------------------------------------------
// WMMA helpers (CDNA5 wave32, v_wmma_f32_16x16x32_f16)
// ---------------------------------------------------------------------------
__device__ __forceinline__ v8f wmma_f16(v16h a, v16h b, v8f c) {
  // (neg_a, A, neg_b, B, c_mod, C, reuse_a, reuse_b)
  return __builtin_amdgcn_wmma_f32_16x16x32_f16(false, a, false, b, (short)0, c,
                                                false, false);
}

// A fragment: 16x32 (MxK) f16, source row-major with leading dim lda (halves).
// lane<16 -> M=lane; lanes 0-15 hold K=0..15, lanes 16-31 hold K=16..31 (matches B).
// The 8 dwords per lane form two contiguous 16B runs -> two b128 loads.
__device__ __forceinline__ v16h load_frag_a(const _Float16* A, int lda, int m0, int k0) {
  const int lid  = threadIdx.x & 31;
  const int row  = m0 + (lid & 15);
  const int half = lid >> 4;
  const _Float16* p = A + (size_t)row * lda + k0;
  union { v16h v; unsigned int u[8]; } r;
#pragma unroll
  for (int i = 0; i < 8; ++i) {
    const int kk = (i << 1) + (half << 4);
    r.u[i] = *(const unsigned int*)(p + kk);
  }
  return r.v;
}

// B fragment: 32x16 (KxN) f16 from Bt = B^T stored row-major (N x K).
// lanes 0-15 hold K=0..15 for N=lane; lanes 16-31 hold K=16..31.
// -> 16 contiguous halves (32B) per lane = two b128 loads.
__device__ __forceinline__ v16h load_frag_b(const _Float16* Bt, int ldb, int n0, int k0) {
  const int lid = threadIdx.x & 31;
  const int n   = n0 + (lid & 15);
  const int kh  = (lid >> 4) << 4;
  const _Float16* p = Bt + (size_t)n * ldb + k0 + kh;
  union { v16h v; uint4 q[2]; } r;
  r.q[0] = *(const uint4*)(p);
  r.q[1] = *(const uint4*)(p + 8);
  return r.v;
}

// ---------------------------------------------------------------------------
// Generic GEMM: C[M,N] = A(MxK f16 row-major) * Bt(NxK f16 row-major)^T + bias
// One wave per 16x64 output strip. Per K-step: issue A + 4 B fragment loads as
// one clause, then 4 independent WMMAs (partial-wait friendly). N % 64 == 0.
// ---------------------------------------------------------------------------
__global__ __launch_bounds__(256) void gemm_tn(const _Float16* __restrict__ A,
                                               const _Float16* __restrict__ Bt,
                                               const float* __restrict__ bias,
                                               _Float16* __restrict__ Ch,
                                               float* __restrict__ Cf,
                                               int M, int N, int K) {
  const int gid  = blockIdx.x * 8 + (threadIdx.x >> 5);
  const int ngrp = N >> 6;
  if (gid >= (M >> 4) * ngrp) return;
  const int mt = gid / ngrp, ng = gid - mt * ngrp;
  const int n0 = ng << 6;
  const int kblocks = K >> 5;

  v8f acc[4] = {};
  for (int kb = 0; kb < kblocks; ++kb) {
    // branch-free pre-touch of the next K-slab of the streamed operand
    const int kpf = (kb + 1 < kblocks) ? kb + 1 : kb;
    __builtin_prefetch(Bt + (size_t)n0 * K + kpf * 32 + (threadIdx.x & 31) * K, 0, 3);

    const v16h a  = load_frag_a(A, K, mt * 16, kb * 32);
    v16h b0 = load_frag_b(Bt, K, n0 +  0, kb * 32);
    v16h b1 = load_frag_b(Bt, K, n0 + 16, kb * 32);
    v16h b2 = load_frag_b(Bt, K, n0 + 32, kb * 32);
    v16h b3 = load_frag_b(Bt, K, n0 + 48, kb * 32);
    acc[0] = wmma_f16(a, b0, acc[0]);
    acc[1] = wmma_f16(a, b1, acc[1]);
    acc[2] = wmma_f16(a, b2, acc[2]);
    acc[3] = wmma_f16(a, b3, acc[3]);
  }

  const int lid   = threadIdx.x & 31;
  const int nn    = lid & 15;
  const int mbase = mt * 16 + ((lid >> 4) << 3);
#pragma unroll
  for (int tt = 0; tt < 4; ++tt) {
    const int n    = n0 + tt * 16 + nn;
    const float bb = bias ? bias[n] : 0.f;
#pragma unroll
    for (int r = 0; r < 8; ++r) {
      const float val = acc[tt][r] + bb;
      if (Ch) Ch[(size_t)(mbase + r) * N + n] = (_Float16)val;
      else    Cf[(size_t)(mbase + r) * N + n] = val;
    }
  }
}

// ---------------------------------------------------------------------------
// LSTM recurrent step GEMM: G[b,n] = X_t[b,:]*Wih[n,:] + H[b,:]*Whh[n,:]
// M=128, N=2048, K=512+512. 16x64 strip per wave -> 256 waves/side.
// ---------------------------------------------------------------------------
__global__ __launch_bounds__(256) void lstm_step_gemm(
    const _Float16* __restrict__ Xl, const _Float16* __restrict__ Xr,
    const _Float16* __restrict__ Hl, const _Float16* __restrict__ Hr,
    const _Float16* __restrict__ WihL, const _Float16* __restrict__ WhhL,
    const _Float16* __restrict__ WihR, const _Float16* __restrict__ WhhR,
    float* __restrict__ GtL, float* __restrict__ GtR, int t) {
  const int side = blockIdx.z;
  const _Float16* X   = side ? Xr : Xl;
  const _Float16* H   = side ? Hr : Hl;
  const _Float16* Wih = side ? WihR : WihL;
  const _Float16* Whh = side ? WhhR : WhhL;
  float* Gt           = side ? GtR : GtL;

  const int gid = blockIdx.x * 8 + (threadIdx.x >> 5); // 0..255
  const int mt = gid >> 5;           // 8 M-tiles
  const int n0 = (gid & 31) << 6;    // 32 N-groups of 64
  const _Float16* Xt = X + (size_t)t * BATCH * EMB;

  v8f acc[4] = {};
#pragma unroll 2
  for (int kb = 0; kb < 16; ++kb) {
    const int kpf = (kb + 1 < 16) ? kb + 1 : kb;
    __builtin_prefetch(Wih + (size_t)n0 * EMB + kpf * 32 + (threadIdx.x & 31) * EMB, 0, 3);
    const v16h a  = load_frag_a(Xt, EMB, mt * 16, kb * 32);
    v16h b0 = load_frag_b(Wih, EMB, n0 +  0, kb * 32);
    v16h b1 = load_frag_b(Wih, EMB, n0 + 16, kb * 32);
    v16h b2 = load_frag_b(Wih, EMB, n0 + 32, kb * 32);
    v16h b3 = load_frag_b(Wih, EMB, n0 + 48, kb * 32);
    acc[0] = wmma_f16(a, b0, acc[0]);
    acc[1] = wmma_f16(a, b1, acc[1]);
    acc[2] = wmma_f16(a, b2, acc[2]);
    acc[3] = wmma_f16(a, b3, acc[3]);
  }
#pragma unroll 2
  for (int kb = 0; kb < 16; ++kb) {
    const int kpf = (kb + 1 < 16) ? kb + 1 : kb;
    __builtin_prefetch(Whh + (size_t)n0 * HID + kpf * 32 + (threadIdx.x & 31) * HID, 0, 3);
    const v16h a  = load_frag_a(H, HID, mt * 16, kb * 32);
    v16h b0 = load_frag_b(Whh, HID, n0 +  0, kb * 32);
    v16h b1 = load_frag_b(Whh, HID, n0 + 16, kb * 32);
    v16h b2 = load_frag_b(Whh, HID, n0 + 32, kb * 32);
    v16h b3 = load_frag_b(Whh, HID, n0 + 48, kb * 32);
    acc[0] = wmma_f16(a, b0, acc[0]);
    acc[1] = wmma_f16(a, b1, acc[1]);
    acc[2] = wmma_f16(a, b2, acc[2]);
    acc[3] = wmma_f16(a, b3, acc[3]);
  }

  const int lid   = threadIdx.x & 31;
  const int nn    = lid & 15;
  const int mbase = mt * 16 + ((lid >> 4) << 3);
#pragma unroll
  for (int tt = 0; tt < 4; ++tt) {
    const int n = n0 + tt * 16 + nn;
#pragma unroll
    for (int r = 0; r < 8; ++r)
      Gt[(size_t)(mbase + r) * G4H + n] = acc[tt][r];
  }
}

// ---------------------------------------------------------------------------
// LSTM pointwise: gate nonlinearities, state update, h -> f16 (fp32 VALU)
// ---------------------------------------------------------------------------
__device__ __forceinline__ float sigf(float x) { return 1.f / (1.f + __expf(-x)); }

__global__ __launch_bounds__(256) void lstm_pointwise(
    const float* __restrict__ GtL, const float* __restrict__ GtR,
    const float* __restrict__ bihL, const float* __restrict__ bhhL,
    const float* __restrict__ bihR, const float* __restrict__ bhhR,
    float* __restrict__ Cl, float* __restrict__ Cr,
    _Float16* __restrict__ Hl, _Float16* __restrict__ Hr,
    _Float16* __restrict__ hsl, _Float16* __restrict__ hsr, int t) {
  const int side = blockIdx.z;
  const float* Gt  = side ? GtR : GtL;
  const float* bih = side ? bihR : bihL;
  const float* bhh = side ? bhhR : bhhL;
  float* C   = side ? Cr : Cl;
  _Float16* H  = side ? Hr : Hl;
  _Float16* hs = side ? hsr : hsl;

  const int idx = blockIdx.x * 256 + threadIdx.x; // 0..65535
  const int b = idx >> 9, j = idx & 511;
  const size_t base = (size_t)b * G4H + j;
  const float gi = Gt[base       ] + bih[j       ] + bhh[j       ];
  const float gf = Gt[base +  512] + bih[j +  512] + bhh[j +  512];
  const float gg = Gt[base + 1024] + bih[j + 1024] + bhh[j + 1024];
  const float go = Gt[base + 1536] + bih[j + 1536] + bhh[j + 1536];
  float c = sigf(gf) * C[(size_t)b * HID + j] + sigf(gi) * tanhf(gg);
  float h = sigf(go) * tanhf(c);
  C[(size_t)b * HID + j] = c;
  const _Float16 h16 = (_Float16)h;
  H[(size_t)b * HID + j] = h16;
  hs[((size_t)t * BATCH + b) * HID + j] = h16;
}

// ---------------------------------------------------------------------------
// Attention: one workgroup per (head n, time t). Q,K,V rows = batch (128), d=64.
// S = QK^T/8 via WMMA into LDS, fp32 softmax in place (probs re-stored f16 over
// consumed score rows), O = P V via WMMA from LDS.
// ---------------------------------------------------------------------------
__global__ __launch_bounds__(256) void attention_kernel(
    const _Float16* __restrict__ q, const _Float16* __restrict__ k,
    const _Float16* __restrict__ v, _Float16* __restrict__ att) {
  __shared__ float    S[128 * 128];   // 64 KB scores; reused as f16 probs
  __shared__ _Float16 Vt[ASZ * 128];  // 16 KB V^T (d, b)

  const int n = blockIdx.x >> 8;      // head
  const int t = blockIdx.x & 255;     // time
  const int wid = threadIdx.x >> 5;   // wave 0..7
  const int lid = threadIdx.x & 31;
  const size_t base = (size_t)t * BATCH * (NH * ASZ) + n * ASZ;

  // stage V^T
  for (int i = threadIdx.x; i < ASZ * 128; i += 256) {
    const int d = i >> 7, bp = i & 127;
    Vt[d * 128 + bp] = v[base + (size_t)bp * (NH * ASZ) + d];
  }

  // scores: wave wid owns M-tile wid; Q fragments hoisted and reused 8x
  {
    const v16h qa0 = load_frag_a(q + base, NH * ASZ, wid * 16, 0);
    const v16h qa1 = load_frag_a(q + base, NH * ASZ, wid * 16, 32);
    for (int ntile = 0; ntile < 8; ++ntile) {
      v16h kb0 = load_frag_b(k + base, NH * ASZ, ntile * 16, 0);
      v16h kb1 = load_frag_b(k + base, NH * ASZ, ntile * 16, 32);
      v8f acc = {};
      acc = wmma_f16(qa0, kb0, acc);
      acc = wmma_f16(qa1, kb1, acc);
      const int ncol  = ntile * 16 + (lid & 15);
      const int mbase = wid * 16 + ((lid >> 4) << 3);
#pragma unroll
      for (int r = 0; r < 8; ++r)
        S[(mbase + r) * 128 + ncol] = acc[r] * 0.125f; // 1/sqrt(64)
    }
  }
  __syncthreads();

  // softmax per row, probs written f16 in place over the row's floats
  if (threadIdx.x < 128) {
    const int r = threadIdx.x;
    float* row = S + r * 128;
    float mx = -1e30f;
    for (int c = 0; c < 128; ++c) mx = fmaxf(mx, row[c]);
    float sum = 0.f;
    for (int c = 0; c < 128; ++c) sum += __expf(row[c] - mx);
    const float inv = 1.f / sum;
    _Float16* prow = (_Float16*)row;  // clobbers row[c/2], already consumed
    for (int c = 0; c < 128; ++c) {
      const float p = __expf(row[c] - mx) * inv;
      prow[c] = (_Float16)p;
    }
  }
  __syncthreads();

  // O = P(128x128) * V(128x64): P fragments hoisted, reused across 4 N-tiles
  {
    const _Float16* P = (const _Float16*)S; // leading dim 256 halves
    v16h pa[4];
#pragma unroll
    for (int kb = 0; kb < 4; ++kb)
      pa[kb] = load_frag_a(P, 256, wid * 16, kb * 32);
#pragma unroll
    for (int ntile = 0; ntile < 4; ++ntile) {
      v8f acc = {};
#pragma unroll
      for (int kb = 0; kb < 4; ++kb)
        acc = wmma_f16(pa[kb], load_frag_b(Vt, 128, ntile * 16, kb * 32), acc);
      const int d     = ntile * 16 + (lid & 15);
      const int mbase = wid * 16 + ((lid >> 4) << 3);
#pragma unroll
      for (int r = 0; r < 8; ++r)
        att[base + (size_t)(mbase + r) * (NH * ASZ) + d] = (_Float16)acc[r];
    }
  }
}

// ---------------------------------------------------------------------------
// Prep / glue kernels
// ---------------------------------------------------------------------------
__global__ void conv_f16(_Float16* dst, const float* src, int n) {
  const int i = blockIdx.x * 256 + threadIdx.x;
  if (i < n) dst[i] = (_Float16)src[i];
}

// dst[c*rows + r] = (f16) src[r*cols + c]
__global__ void transpose_f16(_Float16* dst, const float* src, int rows, int cols) {
  const int i = blockIdx.x * 256 + threadIdx.x;
  if (i >= rows * cols) return;
  const int r = i / cols, c = i - r * cols;
  dst[(size_t)c * rows + r] = (_Float16)src[i];
}

// Wq (N,H,A) -> WqT[(n*A+d)][h]
__global__ void qkv_transpose(_Float16* dst, const float* src) {
  const int i = blockIdx.x * 256 + threadIdx.x; // 512*512
  const int c = i >> 9, h = i & 511;
  const int n = c >> 6, d = c & 63;
  dst[(size_t)c * HID + h] = (_Float16)src[((size_t)n * HID + h) * ASZ + d];
}

__global__ void embed_gather(const int* __restrict__ seq_l, const int* __restrict__ seq_r,
                             const float* __restrict__ emb_l, const float* __restrict__ emb_r,
                             _Float16* __restrict__ Xl, _Float16* __restrict__ Xr) {
  const int side = blockIdx.z;
  const size_t i = (size_t)blockIdx.x * 256 + threadIdx.x; // TB*512
  const int* seq = side ? seq_r : seq_l;
  const float* emb = side ? emb_r : emb_l;
  _Float16* X = side ? Xr : Xl;
  const size_t tb = i >> 9;
  const int col = (int)(i & 511);
  const int tok = seq[tb];
  X[i] = (_Float16)emb[(size_t)tok * EMB + col];
}

__global__ void init_state(float* Cl, float* Cr, _Float16* Hl, _Float16* Hr) {
  const int i = blockIdx.x * 256 + threadIdx.x; // 65536
  Cl[i] = 0.f; Cr[i] = 0.f;
  Hl[i] = (_Float16)0.f; Hr[i] = (_Float16)0.f;
}

// batch means -> cat16 (T,1024) f16: side0 from olf (fp32), side1 from hsr (f16)
__global__ void mean_kernel(const float* __restrict__ olf,
                            const _Float16* __restrict__ hsr,
                            _Float16* __restrict__ cat16) {
  const int side = blockIdx.z;
  const int i = blockIdx.x * 256 + threadIdx.x; // 256*512
  const int t = i >> 9, h = i & 511;
  float s = 0.f;
  if (side == 0) {
    for (int b = 0; b < BATCH; ++b) s += olf[((size_t)t * BATCH + b) * HID + h];
  } else {
    for (int b = 0; b < BATCH; ++b) s += (float)hsr[((size_t)t * BATCH + b) * HID + h];
  }
  cat16[(size_t)t * 1024 + side * 512 + h] = (_Float16)(s * (1.f / 128.f));
}

__global__ void head_kernel(const float* __restrict__ fc, const float* __restrict__ Wa,
                            const float* __restrict__ ba, float* __restrict__ out) {
  const int t = threadIdx.x; // 256 threads
  float a0 = ba[0], a1 = ba[1];
  for (int k2 = 0; k2 < 64; ++k2) {
    const float f = fc[t * 64 + k2];
    a0 += f * Wa[k2 * 2 + 0];
    a1 += f * Wa[k2 * 2 + 1];
  }
  const float m = fmaxf(a0, a1);
  const float e0 = __expf(a0 - m), e1 = __expf(a1 - m);
  const float inv = 1.f / (e0 + e1);
  out[t * 2 + 0] = e0 * inv;
  out[t * 2 + 1] = e1 * inv;
}

// ---------------------------------------------------------------------------
// Host launcher
// ---------------------------------------------------------------------------
extern "C" void kernel_launch(void* const* d_in, const int* in_sizes, int n_in,
                              void* d_out, int out_size, void* d_ws, size_t ws_size,
                              hipStream_t stream) {
  const int*   seq_l = (const int*)d_in[0];
  const int*   seq_r = (const int*)d_in[1];
  const float* emb_l = (const float*)d_in[2];
  const float* emb_r = (const float*)d_in[3];
  const float* Wih_l = (const float*)d_in[4];
  const float* Whh_l = (const float*)d_in[5];
  const float* bih_l = (const float*)d_in[6];
  const float* bhh_l = (const float*)d_in[7];
  const float* Wih_r = (const float*)d_in[8];
  const float* Whh_r = (const float*)d_in[9];
  const float* bih_r = (const float*)d_in[10];
  const float* bhh_r = (const float*)d_in[11];
  const float* Wq = (const float*)d_in[12]; const float* bq = (const float*)d_in[13];
  const float* Wk = (const float*)d_in[14]; const float* bk = (const float*)d_in[15];
  const float* Wv = (const float*)d_in[16]; const float* bv = (const float*)d_in[17];
  const float* Wo = (const float*)d_in[18]; const float* bo = (const float*)d_in[19];
  const float* Wh = (const float*)d_in[20]; const float* bh = (const float*)d_in[21];
  const float* Wa = (const float*)d_in[22]; const float* ba = (const float*)d_in[23];

  char* w = (char*)d_ws;
  size_t off = 0;
  auto alloc = [&](size_t bytes) -> void* {
    void* p = w + off;
    off = (off + bytes + 255) & ~(size_t)255;
    return p;
  };

  _Float16* Xl  = (_Float16*)alloc((size_t)TBTOT * EMB * 2);
  _Float16* Xr  = (_Float16*)alloc((size_t)TBTOT * EMB * 2);
  _Float16* WihL16 = (_Float16*)alloc((size_t)G4H * EMB * 2);
  _Float16* WhhL16 = (_Float16*)alloc((size_t)G4H * HID * 2);
  _Float16* WihR16 = (_Float16*)alloc((size_t)G4H * EMB * 2);
  _Float16* WhhR16 = (_Float16*)alloc((size_t)G4H * HID * 2);
  _Float16* WqT = (_Float16*)alloc((size_t)512 * 512 * 2);
  _Float16* WkT = (_Float16*)alloc((size_t)512 * 512 * 2);
  _Float16* WvT = (_Float16*)alloc((size_t)512 * 512 * 2);
  _Float16* WoT = (_Float16*)alloc((size_t)512 * 512 * 2);
  _Float16* WhT = (_Float16*)alloc((size_t)64 * 1024 * 2);
  float*    Cl  = (float*)alloc((size_t)BATCH * HID * 4);
  float*    Cr  = (float*)alloc((size_t)BATCH * HID * 4);
  _Float16* Hl  = (_Float16*)alloc((size_t)BATCH * HID * 2);
  _Float16* Hr  = (_Float16*)alloc((size_t)BATCH * HID * 2);
  float*    GtL = (float*)alloc((size_t)BATCH * G4H * 4);
  float*    GtR = (float*)alloc((size_t)BATCH * G4H * 4);
  _Float16* hsl = (_Float16*)alloc((size_t)TBTOT * HID * 2);
  _Float16* hsr = (_Float16*)alloc((size_t)TBTOT * HID * 2);
  _Float16* qbuf = (_Float16*)alloc((size_t)TBTOT * 512 * 2);
  _Float16* kbuf = (_Float16*)alloc((size_t)TBTOT * 512 * 2);
  _Float16* vbuf = (_Float16*)alloc((size_t)TBTOT * 512 * 2);
  _Float16* attb = (_Float16*)alloc((size_t)TBTOT * 512 * 2);
  float*    olf  = (float*)alloc((size_t)TBTOT * HID * 4);
  _Float16* cat16 = (_Float16*)alloc((size_t)T_LEN * 1024 * 2);
  float*    fcf   = (float*)alloc((size_t)T_LEN * 64 * 4);
  (void)ws_size; (void)n_in; (void)in_sizes; (void)out_size;

  // weight conversion / transposition
  const int wblk = (G4H * EMB + 255) / 256;
  conv_f16<<<wblk, 256, 0, stream>>>(WihL16, Wih_l, G4H * EMB);
  conv_f16<<<wblk, 256, 0, stream>>>(WhhL16, Whh_l, G4H * HID);
  conv_f16<<<wblk, 256, 0, stream>>>(WihR16, Wih_r, G4H * EMB);
  conv_f16<<<wblk, 256, 0, stream>>>(WhhR16, Whh_r, G4H * HID);
  qkv_transpose<<<1024, 256, 0, stream>>>(WqT, Wq);
  qkv_transpose<<<1024, 256, 0, stream>>>(WkT, Wk);
  qkv_transpose<<<1024, 256, 0, stream>>>(WvT, Wv);
  transpose_f16<<<1024, 256, 0, stream>>>(WoT, Wo, 512, 512);
  transpose_f16<<<256, 256, 0, stream>>>(WhT, Wh, 1024, 64);

  // embeddings + state init
  embed_gather<<<dim3((TBTOT * EMB) / 256, 1, 2), 256, 0, stream>>>(
      seq_l, seq_r, emb_l, emb_r, Xl, Xr);
  init_state<<<(BATCH * HID) / 256, 256, 0, stream>>>(Cl, Cr, Hl, Hr);

  // sequential LSTM scan (both sides per launch via grid z)
  for (int t = 0; t < T_LEN; ++t) {
    lstm_step_gemm<<<dim3(32, 1, 2), 256, 0, stream>>>(
        Xl, Xr, Hl, Hr, WihL16, WhhL16, WihR16, WhhR16, GtL, GtR, t);
    lstm_pointwise<<<dim3(256, 1, 2), 256, 0, stream>>>(
        GtL, GtR, bih_l, bhh_l, bih_r, bhh_r, Cl, Cr, Hl, Hr, hsl, hsr, t);
  }

  // q/k/v projections: (TB,512) x (512,512); 16x64 strips -> 16384 waves
  const int qkv_blocks = ((TBTOT / 16) * (512 / 64)) / 8; // 2048
  gemm_tn<<<qkv_blocks, 256, 0, stream>>>(hsl, WqT, bq, qbuf, nullptr, TBTOT, 512, 512);
  gemm_tn<<<qkv_blocks, 256, 0, stream>>>(hsl, WkT, bk, kbuf, nullptr, TBTOT, 512, 512);
  gemm_tn<<<qkv_blocks, 256, 0, stream>>>(hsl, WvT, bv, vbuf, nullptr, TBTOT, 512, 512);

  // attention per (head, t)
  attention_kernel<<<NH * T_LEN, 256, 0, stream>>>(qbuf, kbuf, vbuf, attb);

  // output projection -> fp32
  gemm_tn<<<qkv_blocks, 256, 0, stream>>>(attb, WoT, bo, nullptr, olf, TBTOT, 512, 512);

  // batch means -> cat (T,1024) f16
  mean_kernel<<<dim3((T_LEN * 512) / 256, 1, 2), 256, 0, stream>>>(olf, hsr, cat16);

  // fc: (256,1024) x (1024,64) -> fp32 ; 16 waves -> 2 blocks
  gemm_tn<<<2, 256, 0, stream>>>(cat16, WhT, bh, nullptr, fcf, T_LEN, 64, 1024);

  // head: (256,64) x (64,2) + softmax -> d_out
  head_kernel<<<1, 256, 0, stream>>>(fcf, Wa, ba, (float*)d_out);
}
